// TemporalLayer_15530601742654
// MI455X (gfx1250) — compile-verified, matrix operand order checked by reading, not run on hardware
//
#include <hip/hip_runtime.h>

// ---------------------------------------------------------------------------
// Problem constants
// ---------------------------------------------------------------------------
#define N_POINTS 32768
#define N_TYPES  4
#define PER_TYPE 8192
#define HIST     32
#define UDIM     128
#define LDA      136   // padded LDS row stride in halfs (16B-aligned, bank-spread)

typedef __attribute__((ext_vector_type(16))) _Float16 v16h;
typedef __attribute__((ext_vector_type(8)))  _Float16 v8h;
typedef __attribute__((ext_vector_type(8)))  float    v8f;

// ---------------------------------------------------------------------------
// WMMA helpers (CDNA5 v_wmma_f32_16x16x32_f16, wave32)
// A fragment (16x32 f16): lane L -> row = L&15, khalf = L>>4;
//   elements e=0..7  : K = khalf*8 + e
//   elements e=8..15 : K = 16 + khalf*8 + (e-8)
// -> two contiguous 8-half (16B) LDS reads per fragment.
// B fragment uses the mirrored layout; weights staged pre-transposed so the
// same loader applies with row = output column.
// ---------------------------------------------------------------------------
__device__ __forceinline__ v16h ld_frag(const _Float16* tile, int row0, int k0, int lane) {
  const int r  = row0 + (lane & 15);
  const int kh = (lane >> 4) & 1;
  const _Float16* p = tile + r * LDA + k0 + kh * 8;
  v8h lo = *(const v8h*)(p);
  v8h hi = *(const v8h*)(p + 16);
  v16h out;
#pragma unroll
  for (int i = 0; i < 8; ++i) { out[i] = lo[i]; out[i + 8] = hi[i]; }
  return out;
}

__device__ __forceinline__ v8f vzero8() { v8f z = {}; return z; }

// One wave computes a 16x128 strip: C[16 x 128] += A[16 x 128] * Bt^T
// Atile: 16-row strip at arow0 (row-major, LDA);  Bt: 128x128 with
// Bt[n][k] = B[k][n] (row-major, LDA).
__device__ __forceinline__ void gemm_16x128(const _Float16* Atile, int arow0,
                                            const _Float16* Bt, int lane, v8f c[8]) {
#pragma unroll
  for (int ks = 0; ks < UDIM; ks += 32) {
    v16h a = ld_frag(Atile, arow0, ks, lane);
#pragma unroll
    for (int nt = 0; nt < 8; ++nt) {
      v16h b = ld_frag(Bt, nt * 16, ks, lane);
      c[nt] = __builtin_amdgcn_wmma_f32_16x16x32_f16(
          false, a, false, b, (short)0, c[nt], false, false);
    }
  }
}

// ---------------------------------------------------------------------------
// Kernel A: gather X, Q = X @ q_w  (store fp32 residual into d_out),
//           P = Q @ k_w^T          (store fp32 into workspace)
// grid 512 (4 types x 128 tiles of 64 rows), block 128 (4 waves)
// ---------------------------------------------------------------------------
__global__ __launch_bounds__(128) void kA(const float* __restrict__ x,
                                          const int*   __restrict__ idx,
                                          const float* __restrict__ q_w,
                                          const float* __restrict__ k_w,
                                          float* __restrict__ q_out,
                                          float* __restrict__ p_out) {
  const int tid  = threadIdx.x;
  const int lane = tid & 31;
  const int wave = tid >> 5;
  const int t    = blockIdx.x >> 7;
  const int row0 = (blockIdx.x & 127) * 64;

  __shared__ __align__(16) _Float16 Atile[64 * LDA];
  __shared__ __align__(16) _Float16 Bq[UDIM * LDA];  // Bq[n=u][k=e] = q_w[e][u]
  __shared__ __align__(16) _Float16 Bk[UDIM * LDA];  // Bk[n=u][k=v] = k_w[u][v]
  __shared__ int sIdx[64];

  if (tid < 64) sIdx[tid] = idx[t * PER_TYPE + row0 + tid];

  const float* qw = q_w + t * UDIM * UDIM;
  const float* kw = k_w + t * UDIM * UDIM;
#pragma unroll 4
  for (int e = 0; e < UDIM; ++e)   // coalesced read, transpose into LDS
    Bq[tid * LDA + e] = (_Float16)qw[e * UDIM + tid];
#pragma unroll 4
  for (int u = 0; u < UDIM; ++u)   // coalesced read, direct (k_w already N-major)
    Bk[u * LDA + tid] = (_Float16)kw[u * UDIM + tid];
  __syncthreads();

#pragma unroll 4
  for (int r = 0; r < 64; ++r)     // gathered A tile, coalesced per row
    Atile[r * LDA + tid] = (_Float16)x[(size_t)sIdx[r] * UDIM + tid];
  __syncthreads();

  v8f c[8];
#pragma unroll
  for (int nt = 0; nt < 8; ++nt) c[nt] = vzero8();
  gemm_16x128(Atile, wave * 16, Bq, lane, c);

  // C/D layout: lane holds col = nt*16 + (lane&15), rows j + 8*(lane>>4)
  const int colbase = lane & 15;
  const int rbase   = 8 * (lane >> 4);
#pragma unroll
  for (int nt = 0; nt < 8; ++nt) {
    const int col = nt * 16 + colbase;
#pragma unroll
    for (int j = 0; j < 8; ++j) {
      const int   ml = wave * 16 + rbase + j;
      const float qv = c[nt][j];
      q_out[(size_t)(t * PER_TYPE + row0 + ml) * UDIM + col] = qv;   // residual
      Atile[ml * LDA + col] = (_Float16)qv;  // wave-local rows; DS is in-order
    }
  }

#pragma unroll
  for (int nt = 0; nt < 8; ++nt) c[nt] = vzero8();
  gemm_16x128(Atile, wave * 16, Bk, lane, c);

#pragma unroll
  for (int nt = 0; nt < 8; ++nt) {
    const int col = nt * 16 + colbase;
#pragma unroll
    for (int j = 0; j < 8; ++j) {
      const int ml = wave * 16 + rbase + j;
      p_out[(size_t)(t * PER_TYPE + row0 + ml) * UDIM + col] = c[nt][j];
    }
  }
}

// ---------------------------------------------------------------------------
// Kernel B: raw[ent][h] = hist[node][h][:] . p[ent][:]
// grid 32768, block 256 (thread -> h = tid>>3, 16-wide u-chunk, shuffle reduce)
// ---------------------------------------------------------------------------
__global__ __launch_bounds__(256) void kB(const float* __restrict__ hist,
                                          const int*   __restrict__ idx,
                                          const float* __restrict__ p_mat,
                                          float* __restrict__ raw) {
  const int ent  = blockIdx.x;
  const int node = idx[ent];
  const int tid  = threadIdx.x;

  __shared__ float ps[UDIM];
  if (tid < UDIM) ps[tid] = p_mat[(size_t)ent * UDIM + tid];
  __syncthreads();

  const int h  = tid >> 3;
  const int u0 = (tid & 7) * 16;
  const float* hp = hist + ((size_t)node * HIST + h) * UDIM + u0;

  float acc = 0.f;
#pragma unroll
  for (int i = 0; i < 16; i += 4) {
    float4 hv = *(const float4*)(hp + i);
    acc += hv.x * ps[u0 + i]     + hv.y * ps[u0 + i + 1]
         + hv.z * ps[u0 + i + 2] + hv.w * ps[u0 + i + 3];
  }
  acc += __shfl_down(acc, 4, 8);
  acc += __shfl_down(acc, 2, 8);
  acc += __shfl_down(acc, 1, 8);
  if ((tid & 7) == 0) raw[(size_t)ent * HIST + h] = acc;
}

// ---------------------------------------------------------------------------
// Kernel C: softmax stats over the entity axis: per (t,h) max & sum(exp)
// grid 128 (= 4*32), block 256
// ---------------------------------------------------------------------------
__global__ __launch_bounds__(256) void kC(const float* __restrict__ raw,
                                          float* __restrict__ stats) {
  const int t = blockIdx.x >> 5;
  const int h = blockIdx.x & 31;
  const int tid = threadIdx.x;
  __shared__ float red[256];

  float m = -3.402823466e38f;
  for (int n = tid; n < PER_TYPE; n += 256)
    m = fmaxf(m, raw[(size_t)(t * PER_TYPE + n) * HIST + h]);
  red[tid] = m;
  __syncthreads();
  for (int s = 128; s > 0; s >>= 1) {
    if (tid < s) red[tid] = fmaxf(red[tid], red[tid + s]);
    __syncthreads();
  }
  m = red[0];
  __syncthreads();

  float sum = 0.f;
  for (int n = tid; n < PER_TYPE; n += 256)
    sum += __expf(raw[(size_t)(t * PER_TYPE + n) * HIST + h] - m);
  red[tid] = sum;
  __syncthreads();
  for (int s = 128; s > 0; s >>= 1) {
    if (tid < s) red[tid] += red[tid + s];
    __syncthreads();
  }
  if (tid == 0) {
    stats[(t * HIST + h) * 2]     = m;
    stats[(t * HIST + h) * 2 + 1] = red[0];
  }
}

// ---------------------------------------------------------------------------
// Kernel D: hbar[ent][u] = sum_h w[h] * hist[node][h][u],
//           w[h] = exp(raw - m) / s
// grid 32768, block 128 (fully coalesced history reads)
// ---------------------------------------------------------------------------
__global__ __launch_bounds__(128) void kD(const float* __restrict__ hist,
                                          const int*   __restrict__ idx,
                                          const float* __restrict__ raw,
                                          const float* __restrict__ stats,
                                          float* __restrict__ hbar) {
  const int ent  = blockIdx.x;
  const int t    = ent >> 13;
  const int node = idx[ent];
  const int tid  = threadIdx.x;

  __shared__ float w[HIST];
  if (tid < HIST) {
    const float m = stats[(t * HIST + tid) * 2];
    const float s = stats[(t * HIST + tid) * 2 + 1];
    w[tid] = __expf(raw[(size_t)ent * HIST + tid] - m) / s;
  }
  __syncthreads();

  const float* hp = hist + (size_t)node * HIST * UDIM;
  float acc = 0.f;
#pragma unroll 8
  for (int h = 0; h < HIST; ++h) acc += w[h] * hp[h * UDIM + tid];
  hbar[(size_t)ent * UDIM + tid] = acc;
}

// ---------------------------------------------------------------------------
// Kernel E: Agg = Hbar @ v_w ; Out = relu(Agg @ fc_w) + Q (Q lives in d_out)
// grid 512, block 128
// ---------------------------------------------------------------------------
__global__ __launch_bounds__(128) void kE(const float* __restrict__ hbar,
                                          const float* __restrict__ v_w,
                                          const float* __restrict__ fc_w,
                                          float* __restrict__ out) {
  const int tid  = threadIdx.x;
  const int lane = tid & 31;
  const int wave = tid >> 5;
  const int t    = blockIdx.x >> 7;
  const int row0 = (blockIdx.x & 127) * 64;

  __shared__ __align__(16) _Float16 Atile[64 * LDA];
  __shared__ __align__(16) _Float16 Bv[UDIM * LDA];  // Bv[n=u][k=e] = v_w[e][u]
  __shared__ __align__(16) _Float16 Bf[UDIM * LDA];  // Bf[n=v][k=u] = fc_w[u][v]

  const float* vw = v_w  + t * UDIM * UDIM;
  const float* fw = fc_w + t * UDIM * UDIM;
#pragma unroll 4
  for (int e = 0; e < UDIM; ++e) Bv[tid * LDA + e] = (_Float16)vw[e * UDIM + tid];
#pragma unroll 4
  for (int u = 0; u < UDIM; ++u) Bf[tid * LDA + u] = (_Float16)fw[u * UDIM + tid];
#pragma unroll 4
  for (int r = 0; r < 64; ++r)
    Atile[r * LDA + tid] =
        (_Float16)hbar[(size_t)(t * PER_TYPE + row0 + r) * UDIM + tid];
  __syncthreads();

  v8f c[8];
#pragma unroll
  for (int nt = 0; nt < 8; ++nt) c[nt] = vzero8();
  gemm_16x128(Atile, wave * 16, Bv, lane, c);

  const int colbase = lane & 15;
  const int rbase   = 8 * (lane >> 4);
#pragma unroll
  for (int nt = 0; nt < 8; ++nt) {
    const int col = nt * 16 + colbase;
#pragma unroll
    for (int j = 0; j < 8; ++j) {
      const int ml = wave * 16 + rbase + j;
      Atile[ml * LDA + col] = (_Float16)c[nt][j];  // agg -> A operand (wave-local)
    }
  }

#pragma unroll
  for (int nt = 0; nt < 8; ++nt) c[nt] = vzero8();
  gemm_16x128(Atile, wave * 16, Bf, lane, c);

#pragma unroll
  for (int nt = 0; nt < 8; ++nt) {
    const int col = nt * 16 + colbase;
#pragma unroll
    for (int j = 0; j < 8; ++j) {
      const int    ml   = wave * 16 + rbase + j;
      const size_t addr = (size_t)(t * PER_TYPE + row0 + ml) * UDIM + col;
      const float  q    = out[addr];              // residual written by kA
      out[addr] = fmaxf(c[nt][j], 0.f) + q;
    }
  }
}

// ---------------------------------------------------------------------------
// Host-side launcher
// ---------------------------------------------------------------------------
extern "C" void kernel_launch(void* const* d_in, const int* in_sizes, int n_in,
                              void* d_out, int out_size, void* d_ws, size_t ws_size,
                              hipStream_t stream) {
  (void)in_sizes; (void)n_in; (void)out_size; (void)ws_size;
  const float* x    = (const float*)d_in[0];  // (32768,128)
  const float* hist = (const float*)d_in[1];  // (32768,32,128)
  const int*   idx  = (const int*)  d_in[2];  // (4,8192)
  const float* q_w  = (const float*)d_in[3];  // (4,128,128)
  const float* k_w  = (const float*)d_in[4];
  const float* v_w  = (const float*)d_in[5];
  const float* fc_w = (const float*)d_in[6];
  float* out = (float*)d_out;                 // (32768,128); also holds Q mid-pipeline

  // Workspace: P (16MB) | raw (4MB) | stats (1KB); hbar reuses P's region
  float* P     = (float*)d_ws;
  float* raw   = P   + (size_t)N_POINTS * UDIM;
  float* stats = raw + (size_t)N_POINTS * HIST;
  float* hbar  = P;  // P is dead after kB

  kA<<<dim3(N_TYPES * (PER_TYPE / 64)), dim3(128), 0, stream>>>(x, idx, q_w, k_w, out, P);
  kB<<<dim3(N_POINTS), dim3(256), 0, stream>>>(hist, idx, P, raw);
  kC<<<dim3(N_TYPES * HIST), dim3(256), 0, stream>>>(raw, stats);
  kD<<<dim3(N_POINTS), dim3(128), 0, stream>>>(hist, idx, raw, stats, hbar);
  kE<<<dim3(N_TYPES * (PER_TYPE / 64)), dim3(128), 0, stream>>>(hbar, v_w, fc_w, out);
}
